// ConfidenceWeightedSlotAttention_21337397527130
// MI455X (gfx1250) — matrix operand order repaired and checked by reading, not compiled, not run on hardware
//
#include <hip/hip_runtime.h>
#include <hip/hip_bf16.h>
#include <cstdint>
#include <cstddef>

typedef __attribute__((ext_vector_type(16))) _Float16 v16h;
typedef __attribute__((ext_vector_type(8)))  _Float16 v8h;
typedef __attribute__((ext_vector_type(8)))  float    v8f;

#define B_   4096
#define KM_  8          // modalities
#define D_   1024
#define S_   16
#define BK_  (B_*KM_)   // 32768 rows
#define BS_  (B_*S_)    // 65536 rows
#define DH_  512
#define D2_  2048
#define D3_  3072

enum { ACT_NONE = 0, ACT_SIG = 1, ACT_RELU = 2 };
enum { OUT_H = 0, OUT_F = 1, OUT_ADDF = 2 };

// ---------------------------------------------------------------------------
// Pack an f32 weight matrix into f16 "Bt" layout: dst[n*K + k] = W[n][k]
// (trans=1 reads src as [K][N] row-major, i.e. performs the transpose).
// ---------------------------------------------------------------------------
__global__ void pack_w_f16(const float* __restrict__ src, _Float16* __restrict__ dst,
                           int N, int K, int trans) {
  size_t i = (size_t)blockIdx.x * blockDim.x + threadIdx.x;
  size_t tot = (size_t)N * (size_t)K;
  if (i >= tot) return;
  int n = (int)(i / K), k = (int)(i % K);
  float v = trans ? src[(size_t)k * N + n] : src[i];
  dst[i] = (_Float16)v;
}

// ---------------------------------------------------------------------------
// Row LayerNorm over D=1024 f32 rows -> f16 normalized row (and optional raw
// f16 cast of the row). One 256-thread block per row.
// ---------------------------------------------------------------------------
__global__ __launch_bounds__(256) void ln_rows_1024(
    const float* __restrict__ x, const float* __restrict__ g,
    const float* __restrict__ bb, _Float16* __restrict__ outLn,
    _Float16* __restrict__ outRaw) {
  __shared__ float red[256];
  int row = blockIdx.x, tid = threadIdx.x;
  const float* xr = x + (size_t)row * 1024;
  float v[4];
#pragma unroll
  for (int t = 0; t < 4; ++t) v[t] = xr[tid + 256 * t];
  red[tid] = v[0] + v[1] + v[2] + v[3];
  __syncthreads();
  for (int o = 128; o > 0; o >>= 1) { if (tid < o) red[tid] += red[tid + o]; __syncthreads(); }
  float mean = red[0] * (1.0f / 1024.0f);
  __syncthreads();
  float sq = 0.f;
#pragma unroll
  for (int t = 0; t < 4; ++t) { float d = v[t] - mean; sq += d * d; }
  red[tid] = sq;
  __syncthreads();
  for (int o = 128; o > 0; o >>= 1) { if (tid < o) red[tid] += red[tid + o]; __syncthreads(); }
  float rstd = rsqrtf(red[0] * (1.0f / 1024.0f) + 1e-5f);
#pragma unroll
  for (int t = 0; t < 4; ++t) {
    int idx = tid + 256 * t;
    float y = (v[t] - mean) * rstd * g[idx] + bb[idx];
    outLn[(size_t)row * 1024 + idx] = (_Float16)y;
    if (outRaw) outRaw[(size_t)row * 1024 + idx] = (_Float16)v[t];
  }
}

// ---------------------------------------------------------------------------
// slots = slot_mu + exp(slot_log_sigma) * slot_noise   (broadcast over batch)
// ---------------------------------------------------------------------------
__global__ void init_slots(const float* __restrict__ noise, const float* __restrict__ mu,
                           const float* __restrict__ lsig, float* __restrict__ out,
                           size_t total) {
  size_t i = (size_t)blockIdx.x * blockDim.x + threadIdx.x;
  if (i >= total) return;
  size_t r = i & (size_t)(S_ * D_ - 1);  // S*D = 16384, power of two
  out[i] = mu[r] + __expf(lsig[r]) * noise[i];
}

// ---------------------------------------------------------------------------
// Confidence head tail: per row of hbuf[BK,512]: LN -> ReLU -> dot(w2) + b2
// -> sigmoid -> conf[row]
// ---------------------------------------------------------------------------
__global__ __launch_bounds__(256) void conf_finalize(
    const float* __restrict__ hbuf, const float* __restrict__ g,
    const float* __restrict__ bb, const float* __restrict__ w2,
    const float* __restrict__ b2, float* __restrict__ conf) {
  __shared__ float red[256];
  int row = blockIdx.x, tid = threadIdx.x;
  const float* xr = hbuf + (size_t)row * 512;
  float v0 = xr[tid], v1 = xr[tid + 256];
  red[tid] = v0 + v1;
  __syncthreads();
  for (int o = 128; o > 0; o >>= 1) { if (tid < o) red[tid] += red[tid + o]; __syncthreads(); }
  float mean = red[0] * (1.0f / 512.0f);
  __syncthreads();
  float d0 = v0 - mean, d1 = v1 - mean;
  red[tid] = d0 * d0 + d1 * d1;
  __syncthreads();
  for (int o = 128; o > 0; o >>= 1) { if (tid < o) red[tid] += red[tid + o]; __syncthreads(); }
  float rstd = rsqrtf(red[0] * (1.0f / 512.0f) + 1e-5f);
  __syncthreads();
  float h0 = fmaxf(d0 * rstd * g[tid] + bb[tid], 0.f);
  float h1 = fmaxf(d1 * rstd * g[tid + 256] + bb[tid + 256], 0.f);
  red[tid] = h0 * w2[tid] + h1 * w2[tid + 256];
  __syncthreads();
  for (int o = 128; o > 0; o >>= 1) { if (tid < o) red[tid] += red[tid + o]; __syncthreads(); }
  if (tid == 0) conf[row] = 1.f / (1.f + __expf(-(red[0] + b2[0])));
}

// ---------------------------------------------------------------------------
// Generic WMMA GEMM: out = act( A1 @ B1^T [+ A2 @ B2^T] + bias1 [+ bias2] )
// A: [M,K] f16 row-major.  B given as Bt: [N,K] f16 row-major (Bt[n][k]).
// One wave (32 threads) computes a 64x64 f32 tile; K stepped by 32.
// Fragment layouts used here (wave32; contiguous k order, not the ISA table order):
//   A (16x32 f16):  lane<16 -> M=lane,   K = k0+0..15
//                   lane>=16 -> M=lane-16, K = k0+16..31
//   B (32x16 f16):  lane<16 -> N=lane,   K = k0+0..15 (16 contiguous halfs)
//                   lane>=16 -> N=lane-16, K = k0+16..31
//   C/D (16x16 f32): VGPR g, lane L -> M = g + (L<16?0:8), N = L%16
// ---------------------------------------------------------------------------
template <int ACT, int OMODE, bool DUAL>
__global__ __launch_bounds__(32) void gemm_wmma_kernel(
    const _Float16* __restrict__ A1, const _Float16* __restrict__ B1,
    const _Float16* __restrict__ A2, const _Float16* __restrict__ B2,
    const float* __restrict__ bias1, const float* __restrict__ bias2,
    _Float16* __restrict__ outH, float* __restrict__ outF,
    int M, int N, int K) {
  (void)M;
  int lane = threadIdx.x & 31;
  int l16 = lane & 15;
  int kA = (lane < 16) ? 0 : 16;
  int kB = (lane < 16) ? 0 : 16;
  int n0 = blockIdx.x * 64;
  int m0 = blockIdx.y * 64;

  v8f acc[4][4] = {};
  for (int k0 = 0; k0 < K; k0 += 32) {
    v16h a[4], b[4];
#pragma unroll
    for (int i = 0; i < 4; ++i) {
      const _Float16* p = A1 + (size_t)(m0 + i * 16 + l16) * K + (k0 + kA);
      v8h lo = *(const v8h*)p;
      v8h hi = *(const v8h*)(p + 8);
      a[i] = __builtin_shufflevector(lo, hi, 0, 1, 2, 3, 4, 5, 6, 7,
                                             8, 9, 10, 11, 12, 13, 14, 15);
    }
#pragma unroll
    for (int j = 0; j < 4; ++j)
      b[j] = *(const v16h*)(B1 + (size_t)(n0 + j * 16 + l16) * K + (k0 + kB));
#pragma unroll
    for (int i = 0; i < 4; ++i)
#pragma unroll
      for (int j = 0; j < 4; ++j)
        acc[i][j] = __builtin_amdgcn_wmma_f32_16x16x32_f16(
            false, a[i], false, b[j], (short)0, acc[i][j], false, false);
    if (DUAL) {
#pragma unroll
      for (int i = 0; i < 4; ++i) {
        const _Float16* p = A2 + (size_t)(m0 + i * 16 + l16) * K + (k0 + kA);
        v8h lo = *(const v8h*)p;
        v8h hi = *(const v8h*)(p + 8);
        a[i] = __builtin_shufflevector(lo, hi, 0, 1, 2, 3, 4, 5, 6, 7,
                                               8, 9, 10, 11, 12, 13, 14, 15);
      }
#pragma unroll
      for (int j = 0; j < 4; ++j)
        b[j] = *(const v16h*)(B2 + (size_t)(n0 + j * 16 + l16) * K + (k0 + kB));
#pragma unroll
      for (int i = 0; i < 4; ++i)
#pragma unroll
        for (int j = 0; j < 4; ++j)
          acc[i][j] = __builtin_amdgcn_wmma_f32_16x16x32_f16(
              false, a[i], false, b[j], (short)0, acc[i][j], false, false);
    }
  }

  int rbase = m0 + ((lane < 16) ? 0 : 8);
#pragma unroll
  for (int i = 0; i < 4; ++i) {
#pragma unroll
    for (int j = 0; j < 4; ++j) {
      int col = n0 + j * 16 + l16;
      float bs = 0.f;
      if (bias1) bs += bias1[col];
      if (DUAL && bias2) bs += bias2[col];
#pragma unroll
      for (int g = 0; g < 8; ++g) {
        int row = rbase + i * 16 + g;
        float v = acc[i][j][g] + bs;
        if (ACT == ACT_SIG)  v = 1.f / (1.f + __expf(-v));
        if (ACT == ACT_RELU) v = fmaxf(v, 0.f);
        size_t idx = (size_t)row * N + col;
        if (OMODE == OUT_H)      outH[idx] = (_Float16)v;
        else if (OMODE == OUT_F) outF[idx] = v;
        else                     outF[idx] += v;
      }
    }
  }
}

// ---------------------------------------------------------------------------
// GRU n-gate + state update, dual-accumulator WMMA GEMM (N fixed = 1024):
//   p1 = upd @ Wih_n^T + b_ih_n ; p2 = h @ Whh_n^T + b_hh_n
//   n  = tanh(p1 + r*p2) ; slots = (1-z)*n + z*h   (r,z from rz buffer)
// One wave computes a 64x32 tile (4x2 fragments, two accumulators each).
// ---------------------------------------------------------------------------
__global__ __launch_bounds__(32) void gru_n_wmma(
    const _Float16* __restrict__ Au, const _Float16* __restrict__ Bih,
    const _Float16* __restrict__ As, const _Float16* __restrict__ Bhh,
    const float* __restrict__ bih, const float* __restrict__ bhh,
    const _Float16* __restrict__ rz, float* __restrict__ slots, int K) {
  int lane = threadIdx.x & 31;
  int l16 = lane & 15;
  int kA = (lane < 16) ? 0 : 16;
  int kB = (lane < 16) ? 0 : 16;
  int n0 = blockIdx.x * 32;
  int m0 = blockIdx.y * 64;

  v8f acc1[4][2] = {}, acc2[4][2] = {};
  for (int k0 = 0; k0 < K; k0 += 32) {
    v16h au[4], as2[4], b1[2], b2[2];
#pragma unroll
    for (int i = 0; i < 4; ++i) {
      const _Float16* pu = Au + (size_t)(m0 + i * 16 + l16) * K + (k0 + kA);
      const _Float16* ps = As + (size_t)(m0 + i * 16 + l16) * K + (k0 + kA);
      v8h lo = *(const v8h*)pu;
      v8h hi = *(const v8h*)(pu + 8);
      au[i] = __builtin_shufflevector(lo, hi, 0, 1, 2, 3, 4, 5, 6, 7,
                                              8, 9, 10, 11, 12, 13, 14, 15);
      lo = *(const v8h*)ps;
      hi = *(const v8h*)(ps + 8);
      as2[i] = __builtin_shufflevector(lo, hi, 0, 1, 2, 3, 4, 5, 6, 7,
                                               8, 9, 10, 11, 12, 13, 14, 15);
    }
#pragma unroll
    for (int j = 0; j < 2; ++j) {
      b1[j] = *(const v16h*)(Bih + (size_t)(n0 + j * 16 + l16) * K + (k0 + kB));
      b2[j] = *(const v16h*)(Bhh + (size_t)(n0 + j * 16 + l16) * K + (k0 + kB));
    }
#pragma unroll
    for (int i = 0; i < 4; ++i)
#pragma unroll
      for (int j = 0; j < 2; ++j) {
        acc1[i][j] = __builtin_amdgcn_wmma_f32_16x16x32_f16(
            false, au[i], false, b1[j], (short)0, acc1[i][j], false, false);
        acc2[i][j] = __builtin_amdgcn_wmma_f32_16x16x32_f16(
            false, as2[i], false, b2[j], (short)0, acc2[i][j], false, false);
      }
  }

  int rbase = m0 + ((lane < 16) ? 0 : 8);
#pragma unroll
  for (int i = 0; i < 4; ++i) {
#pragma unroll
    for (int j = 0; j < 2; ++j) {
      int col = n0 + j * 16 + l16;
      float b1s = bih[col], b2s = bhh[col];
#pragma unroll
      for (int g = 0; g < 8; ++g) {
        int row = rbase + i * 16 + g;
        size_t irz = (size_t)row * 2048;
        float r = (float)rz[irz + col];
        float z = (float)rz[irz + 1024 + col];
        size_t io = (size_t)row * 1024 + col;
        float h = slots[io];
        float nn = tanhf((acc1[i][j][g] + b1s) + r * (acc2[i][j][g] + b2s));
        slots[io] = (1.f - z) * nn + z * h;
      }
    }
  }
}

// ---------------------------------------------------------------------------
// Per-batch attention: logits = (q.k)/32, softmax over K=8, conf-weight,
// renormalize, updates = cw @ v.  One 256-thread block per batch element.
// ---------------------------------------------------------------------------
__global__ __launch_bounds__(256) void slot_attention(
    const _Float16* __restrict__ q, const _Float16* __restrict__ kmat,
    const _Float16* __restrict__ vmat, const float* __restrict__ conf,
    _Float16* __restrict__ upd) {
  __shared__ _Float16 ks[8 * 1024];
  __shared__ _Float16 vs[8 * 1024];
  __shared__ float lg[128];
  __shared__ float cw[16 * 8];
  __shared__ float cfs[8];
  int b = blockIdx.x, tid = threadIdx.x;
  size_t base = (size_t)b * 8 * 1024;
  for (int i = tid; i < 8192; i += 256) { ks[i] = kmat[base + i]; vs[i] = vmat[base + i]; }
  if (tid < 8) cfs[tid] = conf[b * 8 + tid];
  __syncthreads();
  if (tid < 128) {
    int s = tid >> 3, kk = tid & 7;
    const _Float16* qp = q + ((size_t)b * 16 + s) * 1024;
    float acc = 0.f;
#pragma unroll 8
    for (int i = 0; i < 1024; ++i) acc += (float)qp[i] * (float)ks[kk * 1024 + i];
    lg[tid] = acc * 0.03125f;  // 1/sqrt(1024), TEMP=1
  }
  __syncthreads();
  if (tid < 16) {
    float mx = -1e30f;
    for (int k2 = 0; k2 < 8; ++k2) mx = fmaxf(mx, lg[tid * 8 + k2]);
    float e[8], E = 0.f;
    for (int k2 = 0; k2 < 8; ++k2) { e[k2] = __expf(lg[tid * 8 + k2] - mx); E += e[k2]; }
    float sum = 0.f;
    for (int k2 = 0; k2 < 8; ++k2) { e[k2] = (e[k2] / E) * cfs[k2]; sum += e[k2]; }
    float inv = 1.0f / (sum + 1e-8f);
    for (int k2 = 0; k2 < 8; ++k2) cw[tid * 8 + k2] = e[k2] * inv;
  }
  __syncthreads();
  for (int s = 0; s < 16; ++s) {
    for (int d = tid; d < 1024; d += 256) {
      float u = 0.f;
#pragma unroll
      for (int k2 = 0; k2 < 8; ++k2) u += cw[s * 8 + k2] * (float)vs[k2 * 1024 + d];
      upd[((size_t)b * 16 + s) * 1024 + d] = (_Float16)u;
    }
  }
}

// ---------------------------------------------------------------------------
extern "C" void kernel_launch(void* const* d_in, const int* in_sizes, int n_in,
                              void* d_out, int out_size, void* d_ws, size_t ws_size,
                              hipStream_t stream) {
  (void)in_sizes; (void)n_in; (void)out_size; (void)ws_size;
  const float* modality   = (const float*)d_in[0];
  const float* slot_noise = (const float*)d_in[1];
  const float* slot_mu    = (const float*)d_in[2];
  const float* slot_lsig  = (const float*)d_in[3];
  const float* ln_in_g    = (const float*)d_in[4];
  const float* ln_in_b    = (const float*)d_in[5];
  const float* ln_sl_g    = (const float*)d_in[6];
  const float* ln_sl_b    = (const float*)d_in[7];
  const float* wq         = (const float*)d_in[8];
  const float* wk         = (const float*)d_in[9];
  const float* wv         = (const float*)d_in[10];
  const float* gru_w_ih   = (const float*)d_in[11];
  const float* gru_b_ih   = (const float*)d_in[12];
  const float* gru_w_hh   = (const float*)d_in[13];
  const float* gru_b_hh   = (const float*)d_in[14];
  const float* mlp_w1     = (const float*)d_in[15];
  const float* mlp_b1     = (const float*)d_in[16];
  const float* mlp_w2     = (const float*)d_in[17];
  const float* mlp_b2     = (const float*)d_in[18];
  const float* conf_w1    = (const float*)d_in[19];
  const float* conf_b1    = (const float*)d_in[20];
  const float* conf_ln_g  = (const float*)d_in[21];
  const float* conf_ln_b  = (const float*)d_in[22];
  const float* conf_w2    = (const float*)d_in[23];
  const float* conf_b2    = (const float*)d_in[24];
  float* outF = (float*)d_out;  // [BS, D] slots, used as working state

  // ---- workspace layout -------------------------------------------------
  uint8_t* ws = (uint8_t*)d_ws;
  size_t off = 0;
  auto alloc = [&](size_t bytes) -> void* {
    void* p = ws + off;
    off = (off + bytes + 255) & ~(size_t)255;
    return p;
  };
  _Float16* Xln  = (_Float16*)alloc((size_t)BK_ * D_ * 2);   // 64MB
  _Float16* Xraw = (_Float16*)alloc((size_t)BK_ * D_ * 2);   // 64MB (contiguous after Xln)
  _Float16* kmat = (_Float16*)alloc((size_t)BK_ * D_ * 2);   // 64MB
  _Float16* vmat = (_Float16*)alloc((size_t)BK_ * D_ * 2);   // 64MB
  _Float16* sln  = (_Float16*)alloc((size_t)BS_ * D_ * 2);   // 128MB
  _Float16* sprev= (_Float16*)alloc((size_t)BS_ * D_ * 2);   // 128MB
  _Float16* updb = (_Float16*)alloc((size_t)BS_ * D_ * 2);   // 128MB
  _Float16* rzh1 = (_Float16*)alloc((size_t)BS_ * D2_ * 2);  // 256MB (rz, then mlp hidden)
  float*    hbuf = (float*)alloc((size_t)BK_ * DH_ * 4);     // 64MB
  float*    confb= (float*)alloc((size_t)BK_ * 4);
  _Float16* wqh  = (_Float16*)alloc((size_t)D_ * D_ * 2);
  _Float16* wkh  = (_Float16*)alloc((size_t)D_ * D_ * 2);
  _Float16* wvh  = (_Float16*)alloc((size_t)D_ * D_ * 2);
  _Float16* wih  = (_Float16*)alloc((size_t)D3_ * D_ * 2);
  _Float16* whh  = (_Float16*)alloc((size_t)D3_ * D_ * 2);
  _Float16* w1t  = (_Float16*)alloc((size_t)D2_ * D_ * 2);
  _Float16* w2t  = (_Float16*)alloc((size_t)D_ * D2_ * 2);
  _Float16* cw1t = (_Float16*)alloc((size_t)DH_ * D_ * 2);
  // q overlays Xln+Xraw (128MB, free after the k/v GEMMs)
  _Float16* qbuf = Xln;
  _Float16* rz = rzh1;
  _Float16* h1 = rzh1;

  // ---- pack weights to f16 (Bt layout = [out_features][in_features]) ----
  auto packN = [&](const float* src, _Float16* dst, int N, int K, int tr) {
    size_t tot = (size_t)N * K;
    pack_w_f16<<<dim3((unsigned)((tot + 255) / 256)), 256, 0, stream>>>(src, dst, N, K, tr);
  };
  packN(wq, wqh, D_, D_, 0);
  packN(wk, wkh, D_, D_, 0);
  packN(wv, wvh, D_, D_, 0);
  packN(gru_w_ih, wih, D3_, D_, 0);
  packN(gru_w_hh, whh, D3_, D_, 0);
  packN(mlp_w1, w1t, D2_, D_, 1);   // (D,2D) -> [2D][D]
  packN(mlp_w2, w2t, D_, D2_, 1);   // (2D,D) -> [D][2D]
  packN(conf_w1, cw1t, DH_, D_, 1); // (D,D/2) -> [D/2][D]

  // ---- inputs: LN + raw f16 cast ---------------------------------------
  ln_rows_1024<<<BK_, 256, 0, stream>>>(modality, ln_in_g, ln_in_b, Xln, Xraw);

  // ---- confidence head: hbuf = Xraw @ conf_w1 + b1 ; finalize ----------
  gemm_wmma_kernel<ACT_NONE, OUT_F, false><<<dim3(DH_ / 64, BK_ / 64), 32, 0, stream>>>(
      Xraw, cw1t, nullptr, nullptr, conf_b1, nullptr, nullptr, hbuf, BK_, DH_, D_);
  conf_finalize<<<BK_, 256, 0, stream>>>(hbuf, conf_ln_g, conf_ln_b, conf_w2, conf_b2, confb);

  // ---- k, v projections -------------------------------------------------
  gemm_wmma_kernel<ACT_NONE, OUT_H, false><<<dim3(D_ / 64, BK_ / 64), 32, 0, stream>>>(
      Xln, wkh, nullptr, nullptr, nullptr, nullptr, kmat, nullptr, BK_, D_, D_);
  gemm_wmma_kernel<ACT_NONE, OUT_H, false><<<dim3(D_ / 64, BK_ / 64), 32, 0, stream>>>(
      Xln, wvh, nullptr, nullptr, nullptr, nullptr, vmat, nullptr, BK_, D_, D_);

  // ---- slots init -------------------------------------------------------
  {
    size_t total = (size_t)BS_ * D_;
    init_slots<<<dim3((unsigned)((total + 255) / 256)), 256, 0, stream>>>(
        slot_noise, slot_mu, slot_lsig, outF, total);
  }

  // ---- slot attention iterations ---------------------------------------
  for (int it = 0; it < 3; ++it) {
    // slots_ln = LN(slots); slots_prev = f16(slots)
    ln_rows_1024<<<BS_, 256, 0, stream>>>(outF, ln_sl_g, ln_sl_b, sln, sprev);
    // q = slots_ln @ wq^T
    gemm_wmma_kernel<ACT_NONE, OUT_H, false><<<dim3(D_ / 64, BS_ / 64), 32, 0, stream>>>(
        sln, wqh, nullptr, nullptr, nullptr, nullptr, qbuf, nullptr, BS_, D_, D_);
    // attention + confidence weighting -> updates
    slot_attention<<<B_, 256, 0, stream>>>(qbuf, kmat, vmat, confb, updb);
    // rz = sigmoid(upd@Wih_rz^T + h@Whh_rz^T + b_ih_rz + b_hh_rz)
    gemm_wmma_kernel<ACT_SIG, OUT_H, true><<<dim3(D2_ / 64, BS_ / 64), 32, 0, stream>>>(
        updb, wih, sprev, whh, gru_b_ih, gru_b_hh, rz, nullptr, BS_, D2_, D_);
    // n-gate + GRU state update -> slots (in place in d_out)
    gru_n_wmma<<<dim3(D_ / 32, BS_ / 64), 32, 0, stream>>>(
        updb, wih + (size_t)D2_ * D_, sprev, whh + (size_t)D2_ * D_,
        gru_b_ih + D2_, gru_b_hh + D2_, rz, outF, D_);
    // sn = LN(slots)
    ln_rows_1024<<<BS_, 256, 0, stream>>>(outF, ln_sl_g, ln_sl_b, sln, nullptr);
    // h1 = relu(sn @ mlp_w1 + b1)
    gemm_wmma_kernel<ACT_RELU, OUT_H, false><<<dim3(D2_ / 64, BS_ / 64), 32, 0, stream>>>(
        sln, w1t, nullptr, nullptr, mlp_b1, nullptr, h1, nullptr, BS_, D2_, D_);
    // slots += h1 @ mlp_w2 + b2
    gemm_wmma_kernel<ACT_NONE, OUT_ADDF, false><<<dim3(D_ / 64, BS_ / 64), 32, 0, stream>>>(
        h1, w2t, nullptr, nullptr, mlp_b2, nullptr, nullptr, outF, BS_, D_, D2_);
  }
}
